// LidarToBEV_84645215470111
// MI455X (gfx1250) — compile-verified, matrix operand order checked by reading, not run on hardware
//
#include <hip/hip_runtime.h>
#include <hip/hip_bf16.h>

typedef __attribute__((ext_vector_type(16))) _Float16 v16h;
typedef __attribute__((ext_vector_type(8)))  float    v8f;
typedef int g4i __attribute__((vector_size(16)));   /* matches builtin param type exactly */

#define AS1 __attribute__((address_space(1)))
#define AS3 __attribute__((address_space(3)))

#if __has_builtin(__builtin_amdgcn_global_load_async_to_lds_b128) && \
    __has_builtin(__builtin_amdgcn_s_wait_asynccnt)
#define HAVE_ASYNC 1
#else
#define HAVE_ASYNC 0
#endif

#define BEVH 512
#define BEVW 512
#define HW   (BEVH*BEVW)        /* 262144 */
#define NPTS 200000
#define EPS  1e-5f

/* workspace layout (bytes) */
#define OFF_BEV  0u                          /* 5*HW u32   = 5,242,880  */
#define OFF_A1   (5u*1024u*1024u)            /* HW*32 f16  = 16,777,216 */
#define OFF_A2   (21u*1024u*1024u)           /* HW*64 f16  = 33,554,432 */
#define OFF_ST1  55574528u                   /* 64 f32 */
#define OFF_ST2  (OFF_ST1 + 256u)            /* 128 f32 */
#define OFF_WP2  (OFF_ST1 + 512u)            /* 9*4*32*16 f16 = 36,864B (32B aligned) */
#define OFF_WP3  (OFF_WP2 + 73728u)          /* 2*8*32*16 f16 = 16,384B */

/* ---- order-preserving float<->uint encoding for atomic max (init = enc(0)=0x80000000) ---- */
__device__ __forceinline__ unsigned enc_f32(float f) {
    unsigned u = __float_as_uint(f);
    return (u & 0x80000000u) ? ~u : (u | 0x80000000u);
}
__device__ __forceinline__ float dec_u32(unsigned u) {
    unsigned b = (u & 0x80000000u) ? (u & 0x7FFFFFFFu) : ~u;
    return __uint_as_float(b);
}

/* ---------------- init encoded BEV grid ---------------- */
__global__ void k_init(unsigned* __restrict__ bev) {
    unsigned i = blockIdx.x * 256u + threadIdx.x;
    if (i < 5u * HW) bev[i] = 0x80000000u;   /* enc(0.0f) */
}

/* ---------------- point scatter (atomic max, order-independent => deterministic) ------- */
__global__ void k_scatter(const float* __restrict__ pts,
                          const float* __restrict__ bins,
                          unsigned* __restrict__ bev) {
    int i = blockIdx.x * 256 + threadIdx.x;
    if (i >= NPTS) return;
    float x = pts[i*4+0], y = pts[i*4+1], z = pts[i*4+2], it = pts[i*4+3];
    int xi = (int)((x + 51.2f) * 5.0f);  xi = min(max(xi, 0), BEVW-1);
    int yi = (int)((y + 51.2f) * 5.0f);  yi = min(max(yi, 0), BEVH-1);
    int h = (bins[0] < z) + (bins[1] < z) + (bins[2] < z) + (bins[3] < z);
    h = min(h, 3);
    atomicMax(&bev[h*HW + yi*BEVW + xi], enc_f32(z + 2.0f));
    atomicMax(&bev[4*HW + yi*BEVW + xi], enc_f32(it));
}

/* ---------------- weight pre-swizzle into per-lane WMMA B-register image ----------------
   element i of a lane's v16h maps to K = (i&7) + 16*(i>>3) + 8*(lane>>4), N = nt*16+(lane&15) */
__global__ void k_prep2(const float* __restrict__ w, _Float16* __restrict__ wp) {
    int idx = blockIdx.x * 256 + threadIdx.x;          /* 9*4*32*16 = 18432 */
    if (idx >= 9*4*32*16) return;
    int i    = idx & 15;
    int lane = (idx >> 4) & 31;
    int nt   = (idx >> 9) & 3;
    int tap  = idx >> 11;
    int ky = tap / 3, kx = tap % 3;
    int n = nt*16 + (lane & 15);
    int k = (i & 7) + ((i >> 3) << 4) + ((lane >> 4) << 3);
    wp[idx] = (_Float16)w[((n*32 + k)*3 + ky)*3 + kx];  /* OIHW (64,32,3,3) */
}
__global__ void k_prep3(const float* __restrict__ w, _Float16* __restrict__ wp) {
    int idx = blockIdx.x * 256 + threadIdx.x;          /* 2*8*32*16 = 8192 */
    if (idx >= 2*8*32*16) return;
    int i    = idx & 15;
    int lane = (idx >> 4) & 31;
    int nt   = (idx >> 9) & 7;
    int ks   = idx >> 12;
    int n = nt*16 + (lane & 15);
    int k = ks*32 + (i & 7) + ((i >> 3) << 4) + ((lane >> 4) << 3);
    wp[idx] = (_Float16)w[n*64 + k];                   /* (128,64,1,1) */
}

/* ---------------- conv1: 5->32, 3x3 direct (tiny K, VALU), NHWC f16 out ---------------- */
__global__ __launch_bounds__(256) void k_conv1(const unsigned* __restrict__ bev,
                                               const float* __restrict__ w,
                                               const float* __restrict__ bias,
                                               _Float16* __restrict__ act1) {
    int pix = blockIdx.x * 256 + threadIdx.x;
    int y = pix >> 9, x = pix & 511;
    float in[45];
    #pragma unroll
    for (int ky = 0; ky < 3; ky++) {
        #pragma unroll
        for (int kx = 0; kx < 3; kx++) {
            int py = y + ky - 1, px = x + kx - 1;
            bool ok = (py >= 0) && (py < BEVH) && (px >= 0) && (px < BEVW);
            int t = ky*3 + kx;
            #pragma unroll
            for (int c = 0; c < 5; c++)
                in[c*9 + t] = ok ? dec_u32(bev[c*HW + py*BEVW + px]) : 0.0f;
        }
    }
    _Float16* op = act1 + (size_t)pix * 32;
    #pragma unroll 4
    for (int co = 0; co < 32; co++) {
        float acc = bias[co];
        #pragma unroll
        for (int c = 0; c < 5; c++)
            #pragma unroll
            for (int t = 0; t < 9; t++)
                acc = fmaf(in[c*9 + t], w[co*45 + c*9 + t], acc);
        op[co] = (_Float16)acc;
    }
}

/* ---------------- per-channel batch stats (deterministic tree reduce) ---------------- */
__global__ __launch_bounds__(256) void k_stats(const _Float16* __restrict__ act,
                                               float* __restrict__ stats, int C) {
    int ch = blockIdx.x, tid = threadIdx.x;
    float s = 0.f, q = 0.f;
    for (int p = tid; p < HW; p += 256) {
        float v = (float)act[(size_t)p * C + ch];
        s += v; q += v * v;
    }
    __shared__ float ls[256], lq[256];
    ls[tid] = s; lq[tid] = q;
    __syncthreads();
    for (int off = 128; off > 0; off >>= 1) {
        if (tid < off) { ls[tid] += ls[tid+off]; lq[tid] += lq[tid+off]; }
        __syncthreads();
    }
    if (tid == 0) { stats[2*ch] = ls[0]; stats[2*ch+1] = lq[0]; }
}

/* ---------------- fused BN(batch-stats)+ReLU, f16 in place ---------------- */
__global__ __launch_bounds__(256) void k_bnrelu(_Float16* __restrict__ act,
                                                const float* __restrict__ stats,
                                                const float* __restrict__ g,
                                                const float* __restrict__ b, int C) {
    __shared__ float sc[128], sh[128];
    int tid = threadIdx.x;
    if (tid < C) {
        float m   = stats[2*tid] * (1.0f/HW);
        float var = stats[2*tid+1] * (1.0f/HW) - m*m;
        float s   = g[tid] * rsqrtf(var + EPS);
        sc[tid] = s;
        sh[tid] = b[tid] - m*s;
    }
    __syncthreads();
    int pix = blockIdx.x * 256 + tid;
    _Float16* p = act + (size_t)pix * C;
    for (int c = 0; c < C; c++) {
        float v = (float)p[c] * sc[c] + sh[c];
        p[c] = (_Float16)(v > 0.f ? v : 0.f);
    }
}

/* ---------------- conv2: 32->64, 3x3 implicit-GEMM, WMMA f16 -------------------------
   Block = one row y, 128 x-pixels (8 waves x 16 px). 3 activation rows staged in LDS via
   async global->LDS copies (ASYNCcnt); halo zeroed; WMMA loop branch-free (EXEC all 1s). */
__global__ __launch_bounds__(256) void k_conv2(const _Float16* __restrict__ act1,
                                               const _Float16* __restrict__ wp,
                                               const float* __restrict__ bias,
                                               _Float16* __restrict__ act2) {
    __shared__ _Float16 la[3 * 132 * 32];        /* 25,344 B: rows y-1..y+1, cols px0-1..px0+130 */
    int tid  = threadIdx.x;
    int lane = tid & 31;
    int wave = tid >> 5;
    int y      = blockIdx.x >> 2;
    int x0_blk = (blockIdx.x & 3) << 7;          /* 0,128,256,384 */
    int m  = lane & 15;
    int hi = lane >> 4;
    int kb = hi * 8;

    if (y + 1 < BEVH)
        __builtin_prefetch(act1 + ((size_t)((y+1)*BEVW + x0_blk)) * 32, 0, 1);

    /* zero only potentially-OOB halo entries */
    uint4 zz = make_uint4(0,0,0,0);
    uint4* l4 = (uint4*)la;
    if (y == 0)        for (int i = tid; i < 528; i += 256) l4[i] = zz;              /* row 0  */
    if (y == BEVH-1)   for (int i = tid; i < 528; i += 256) l4[1056 + i] = zz;       /* row 2  */
    if (x0_blk == 0)         { if (tid < 12) l4[(tid/4)*528 + (tid&3)] = zz; }       /* col 0  */
    if (x0_blk + 128 == BEVW){ if (tid < 12) l4[(tid/4)*528 + 129*4 + (tid&3)] = zz; }/* col 129 */

    /* stage valid region: 3 rows x 130 cols x 4 chunks of 16B */
    for (int t = tid; t < 3*130*4; t += 256) {
        int r   = t / 520;
        int rem = t - r*520;
        int c   = rem >> 2;
        int q   = rem & 3;
        int py = y - 1 + r;
        int px = x0_blk - 1 + c;
        if (py >= 0 && py < BEVH && px >= 0 && px < BEVW) {
            const _Float16* g = act1 + ((size_t)(py*BEVW + px))*32 + q*8;
            _Float16*       l = la + (r*132 + c)*32 + q*8;
#if HAVE_ASYNC
            __builtin_amdgcn_global_load_async_to_lds_b128(
                (AS1 g4i*)g, (AS3 g4i*)l, 0, 0);
#else
            *(uint4*)l = *(const uint4*)g;
#endif
        }
    }
#if HAVE_ASYNC
    __builtin_amdgcn_s_wait_asynccnt(0);
#endif
    __syncthreads();

    v8f acc[4] = {};
    #pragma unroll
    for (int ky = 0; ky < 3; ky++) {
        #pragma unroll
        for (int kx = 0; kx < 3; kx++) {
            int col = wave*16 + kx + m;                      /* in [0,129] always */
            const _Float16* p = la + ((size_t)(ky*132 + col))*32 + kb;
            union { v16h h; uint4 u[2]; } af;
            af.u[0] = *(const uint4*)p;                      /* ds_load_b128 */
            af.u[1] = *(const uint4*)(p + 16);
            int tap = ky*3 + kx;
            const _Float16* bb = wp + ((size_t)(tap*128 + lane)) * 16;
            #pragma unroll
            for (int nt = 0; nt < 4; nt++) {
                v16h b = *(const v16h*)(bb + nt*512);
                acc[nt] = __builtin_amdgcn_wmma_f32_16x16x32_f16(
                    false, af.h, false, b, (short)0, acc[nt], false, false);
            }
        }
    }
    int x0 = x0_blk + wave*16;
    #pragma unroll
    for (int nt = 0; nt < 4; nt++) {
        int ch = nt*16 + m;
        float bv = bias[ch];
        #pragma unroll
        for (int j = 0; j < 8; j++) {
            int pix = y*BEVW + x0 + j + hi*8;
            act2[(size_t)pix*64 + ch] = (_Float16)(acc[nt][j] + bv);
        }
    }
}

/* ---------------- conv3: 64->128, 1x1 GEMM, WMMA f16, vectorized NCHW f32 stores ------ */
__global__ __launch_bounds__(256) void k_conv3(const _Float16* __restrict__ act2,
                                               const _Float16* __restrict__ wp,
                                               const float* __restrict__ bias,
                                               float* __restrict__ out) {
    int lane = threadIdx.x & 31;
    int wave = threadIdx.x >> 5;
    int tile = blockIdx.x * 8 + wave;
    int p0 = tile * 16;
    int m  = lane & 15;
    int hi = lane >> 4;
    int kb = hi * 8;

    const _Float16* base = act2 + (size_t)(p0 + m) * 64;
    union { v16h h; uint4 u[2]; } a[2];
    #pragma unroll
    for (int ks = 0; ks < 2; ks++) {
        a[ks].u[0] = *(const uint4*)(base + ks*32 + kb);
        a[ks].u[1] = *(const uint4*)(base + ks*32 + kb + 16);
    }
    v8f acc[8] = {};
    #pragma unroll
    for (int ks = 0; ks < 2; ks++) {
        const _Float16* bb = wp + ((size_t)(ks*256 + lane)) * 16;
        #pragma unroll
        for (int nt = 0; nt < 8; nt++) {
            v16h b = *(const v16h*)(bb + nt*512);
            acc[nt] = __builtin_amdgcn_wmma_f32_16x16x32_f16(
                false, a[ks].h, false, b, (short)0, acc[nt], false, false);
        }
    }
    #pragma unroll
    for (int nt = 0; nt < 8; nt++) {
        int ch = nt*16 + m;
        float bv = bias[ch];
        float* op = out + (size_t)ch*HW + p0 + hi*8;
        float4 lo = make_float4(acc[nt][0]+bv, acc[nt][1]+bv, acc[nt][2]+bv, acc[nt][3]+bv);
        float4 hi4= make_float4(acc[nt][4]+bv, acc[nt][5]+bv, acc[nt][6]+bv, acc[nt][7]+bv);
        *(float4*)op       = lo;     /* 32B contiguous per lane -> global_store_b128 */
        *(float4*)(op + 4) = hi4;
    }
}

extern "C" void kernel_launch(void* const* d_in, const int* in_sizes, int n_in,
                              void* d_out, int out_size, void* d_ws, size_t ws_size,
                              hipStream_t stream) {
    const float* points = (const float*)d_in[0];
    const float* hbins  = (const float*)d_in[1];
    const float* w1  = (const float*)d_in[2];
    const float* b1  = (const float*)d_in[3];
    const float* g1  = (const float*)d_in[4];
    const float* bb1 = (const float*)d_in[5];
    const float* w2  = (const float*)d_in[6];
    const float* b2  = (const float*)d_in[7];
    const float* g2  = (const float*)d_in[8];
    const float* bb2 = (const float*)d_in[9];
    const float* w3  = (const float*)d_in[10];
    const float* b3  = (const float*)d_in[11];

    char* ws = (char*)d_ws;
    unsigned*  bev = (unsigned*)(ws + OFF_BEV);
    _Float16*  a1  = (_Float16*)(ws + OFF_A1);
    _Float16*  a2  = (_Float16*)(ws + OFF_A2);
    float*     st1 = (float*)(ws + OFF_ST1);
    float*     st2 = (float*)(ws + OFF_ST2);
    _Float16*  wp2 = (_Float16*)(ws + OFF_WP2);
    _Float16*  wp3 = (_Float16*)(ws + OFF_WP3);
    float*     out = (float*)d_out;

    k_init   <<<5120, 256, 0, stream>>>(bev);
    k_scatter<<<(NPTS+255)/256, 256, 0, stream>>>(points, hbins, bev);
    k_prep2  <<<72, 256, 0, stream>>>(w2, wp2);
    k_prep3  <<<32, 256, 0, stream>>>(w3, wp3);
    k_conv1  <<<1024, 256, 0, stream>>>(bev, w1, b1, a1);
    k_stats  <<<32, 256, 0, stream>>>(a1, st1, 32);
    k_bnrelu <<<1024, 256, 0, stream>>>(a1, st1, g1, bb1, 32);
    k_conv2  <<<2048, 256, 0, stream>>>(a1, wp2, b2, a2);
    k_stats  <<<64, 256, 0, stream>>>(a2, st2, 64);
    k_bnrelu <<<1024, 256, 0, stream>>>(a2, st2, g2, bb2, 64);
    k_conv3  <<<2048, 256, 0, stream>>>(a2, wp3, b3, out);
}
